// MultiHeadLatentAttention_61117384622339
// MI455X (gfx1250) — matrix operand order repaired and checked by reading, not compiled, not run on hardware
//
#include <hip/hip_runtime.h>
#include <hip/hip_bf16.h>
#include <cstdint>

typedef __attribute__((ext_vector_type(16))) _Float16 v16h;
typedef __attribute__((ext_vector_type(8)))  _Float16 v8h;
typedef __attribute__((ext_vector_type(4)))  _Float16 v4h;
typedef __attribute__((ext_vector_type(8)))  float    v8f;

#define S_LEN 2048
#define H_NUM 16
#define D_LAT 32
#define BATCH 2
#define ROWS  (BATCH * S_LEN)   // 4096

// Build a v16h A/B fragment from two contiguous 16-byte chunks.
static __device__ inline v16h ld16h(const _Float16* p0, const _Float16* p1) {
  v8h a = *(const v8h*)p0;
  v8h b = *(const v8h*)p1;
  v16h r;
#pragma unroll
  for (int i = 0; i < 8; i++) { r[i] = a[i]; r[i + 8] = b[i]; }
  return r;
}

// CDNA5 async global->LDS copy (16B per lane), tracked by ASYNCcnt.
// Low 32 bits of a flat shared-aperture pointer are the LDS byte offset.
static __device__ inline void async_g2l_b128(const void* gptr, void* lptr) {
  uint32_t lds  = (uint32_t)(uintptr_t)lptr;
  uint64_t gadr = (uint64_t)(uintptr_t)gptr;
  asm volatile("global_load_async_to_lds_b128 %0, %1, off"
               :: "v"(lds), "v"(gadr) : "memory");
}
static __device__ inline void wait_async0() {
  asm volatile("s_wait_asynccnt 0x0" ::: "memory");
}
// Async ops complete in order per wave: <=3 outstanding means the previous
// group of 3 (current tile) has landed while the next tile's DMA is in flight.
static __device__ inline void wait_async3() {
  asm volatile("s_wait_asynccnt 0x3" ::: "memory");
}

// ---------------------------------------------------------------- converters
__global__ __launch_bounds__(256) void cvt_f32_f16(const float* __restrict__ in,
                                                   _Float16* __restrict__ out, int n) {
  int i = (blockIdx.x * 256 + threadIdx.x) * 4;
  if (i < n) {
    float4 v = *(const float4*)(in + i);
    v4h o;
    o[0] = (_Float16)v.x; o[1] = (_Float16)v.y;
    o[2] = (_Float16)v.z; o[3] = (_Float16)v.w;
    *(v4h*)(out + i) = o;
  }
}

// Wt[n][k] = W[k][nsrc]; qsub=1 gathers the first 32 columns of each 64-wide head.
__global__ __launch_bounds__(256) void transpose_w(const float* __restrict__ W,
                                                   _Float16* __restrict__ Wt,
                                                   int K, int Nsrc, int Nout, int qsub) {
  int idx = blockIdx.x * 256 + threadIdx.x;
  if (idx >= Nout * K) return;
  int n = idx / K, k = idx - n * K;
  int nsrc = qsub ? ((n >> 5) * 64 + (n & 31)) : n;
  Wt[(size_t)n * K + k] = (_Float16)W[(size_t)k * Nsrc + nsrc];
}

// ---------------------------------------------------------------- f16 GEMM
// C(M,N) = A(M,K) * Bt(N,K)^T + bias.
// Double-buffered async global->LDS staging (ASYNCcnt partial waits).
// mode: 0 = f32 row-major, 1 = f16 row-major,
//       2 = f16 scatter -> (B,H,S,32), 3 = f16 scatter -> (B,H,32,S)
// bias_mode: 0 = bias[gc], 1 = bias[(gc/32)*64 + gc%32] (Q-lat column map)
__global__ __launch_bounds__(256) void gemm_f16(const _Float16* __restrict__ A,
                                                const _Float16* __restrict__ Bt,
                                                const float* __restrict__ bias,
                                                void* __restrict__ Cout,
                                                int M, int N, int K, int mode, int bias_mode) {
  __shared__ _Float16 As[2][64][32];
  __shared__ _Float16 Bs[2][128][32];

  const int m0 = blockIdx.x * 64, n0 = blockIdx.y * 128;
  const int tid = threadIdx.x;
  const int w = tid >> 5, l = tid & 31;
  const int wm = w & 1, wn = w >> 1;
  const int l15 = l & 15;
  const int ks = (l < 16) ? 0 : 16;
  const int kb = (l < 16) ? 0 : 16;
  const int rloc = (l < 16) ? 0 : 8;
  const int arow = tid >> 2, acb = (tid & 3) * 8;
  const int brow = tid >> 1, bcb = (tid & 1) * 16;

  const _Float16* ag = A  + (size_t)(m0 + arow) * K + acb;
  const _Float16* bg = Bt + (size_t)(n0 + brow) * K + bcb;

  v8f acc[2][2] = {};

  const int nk = K >> 5;
  // prologue: stage tile 0 into buffer 0
  async_g2l_b128(ag,     &As[0][arow][acb]);
  async_g2l_b128(bg,     &Bs[0][brow][bcb]);
  async_g2l_b128(bg + 8, &Bs[0][brow][bcb + 8]);

  for (int i = 0; i < nk; i++) {
    const int cur = i & 1;
    if (i + 1 < nk) {
      const int kt = (i + 1) << 5;
      const int nxt = cur ^ 1;
      async_g2l_b128(ag + kt,     &As[nxt][arow][acb]);
      async_g2l_b128(bg + kt,     &Bs[nxt][brow][bcb]);
      async_g2l_b128(bg + kt + 8, &Bs[nxt][brow][bcb + 8]);
      wait_async3();   // current tile landed; next tile DMA stays in flight
    } else {
      wait_async0();
    }
    __syncthreads();

    v16h af0 = ld16h(&As[cur][wm * 32 + l15][ks],      &As[cur][wm * 32 + l15][ks + 8]);
    v16h af1 = ld16h(&As[cur][wm * 32 + 16 + l15][ks], &As[cur][wm * 32 + 16 + l15][ks + 8]);
    v16h bf0 = ld16h(&Bs[cur][wn * 32 + l15][kb],      &Bs[cur][wn * 32 + l15][kb + 8]);
    v16h bf1 = ld16h(&Bs[cur][wn * 32 + 16 + l15][kb], &Bs[cur][wn * 32 + 16 + l15][kb + 8]);

    acc[0][0] = __builtin_amdgcn_wmma_f32_16x16x32_f16(false, af0, false, bf0, (short)0, acc[0][0], false, false);
    acc[0][1] = __builtin_amdgcn_wmma_f32_16x16x32_f16(false, af0, false, bf1, (short)0, acc[0][1], false, false);
    acc[1][0] = __builtin_amdgcn_wmma_f32_16x16x32_f16(false, af1, false, bf0, (short)0, acc[1][0], false, false);
    acc[1][1] = __builtin_amdgcn_wmma_f32_16x16x32_f16(false, af1, false, bf1, (short)0, acc[1][1], false, false);
    __syncthreads();
  }

#pragma unroll
  for (int mi = 0; mi < 2; mi++)
#pragma unroll
    for (int ni = 0; ni < 2; ni++)
#pragma unroll
      for (int r = 0; r < 8; r++) {
        int gr = m0 + wm * 32 + mi * 16 + r + rloc;
        int gc = n0 + wn * 32 + ni * 16 + l15;
        float v = acc[mi][ni][r];
        v += bias[bias_mode ? ((gc >> 5) * 64 + (gc & 31)) : gc];
        if (mode == 0) {
          ((float*)Cout)[(size_t)gr * N + gc] = v;
        } else if (mode == 1) {
          ((_Float16*)Cout)[(size_t)gr * N + gc] = (_Float16)v;
        } else {
          int bb = gr >> 11, s = gr & (S_LEN - 1);
          int hh = gc >> 5, d = gc & 31;
          size_t idx = (mode == 2)
              ? ((size_t)(bb * H_NUM + hh) * S_LEN + s) * D_LAT + d
              : ((size_t)(bb * H_NUM + hh) * D_LAT + d) * S_LEN + s;
          ((_Float16*)Cout)[idx] = (_Float16)v;
        }
      }
}

// ---------------------------------------------------------------- attention
// Per block: one (b, h, 64-row q strip). 4 waves, each owns 16 q rows.
// Pass 1: online softmax stats via WMMA score tiles + 16-lane reductions.
// Pass 2: recompute scores, write normalized attn (nontemporal f32),
//         relayout P through LDS into A-fragments, accumulate ctx = P*V.
__global__ __launch_bounds__(128) void mla_attention(const _Float16* __restrict__ Qlat,
                                                     const _Float16* __restrict__ Klat,
                                                     const _Float16* __restrict__ VlatT,
                                                     _Float16* __restrict__ ctx,
                                                     float* __restrict__ attn) {
  __shared__ _Float16 pl[4][16][32];

  const int b = blockIdx.z, h = blockIdx.y;
  const int w = threadIdx.x >> 5, l = threadIdx.x & 31;
  const int l15 = l & 15;
  const int ks = (l < 16) ? 0 : 16;
  const int kb = (l < 16) ? 0 : 16;
  const int rloc = (l < 16) ? 0 : 8;
  const int q0 = blockIdx.x * 64 + w * 16;
  const float scale = 0.17677669529663687f;  // 1/sqrt(32)

  const size_t bh = (size_t)(b * H_NUM + h);
  const _Float16* Qb = Qlat + bh * S_LEN * D_LAT;
  const _Float16* Kb = Klat + bh * S_LEN * D_LAT;
  const _Float16* Vb = VlatT + bh * D_LAT * S_LEN;
  float* attn_b = attn + bh * (size_t)S_LEN * S_LEN;

  const _Float16* qp = Qb + (size_t)(q0 + l15) * D_LAT;
  v16h qf = ld16h(qp + ks, qp + ks + 8);

  float mrow[8], srow[8];
#pragma unroll
  for (int r = 0; r < 8; r++) { mrow[r] = -1e30f; srow[r] = 0.f; }

  // ---- pass 1: row max + sum of exp
  for (int k0 = 0; k0 < S_LEN; k0 += 16) {
    const _Float16* kp = Kb + (size_t)(k0 + l15) * D_LAT + kb;
    v16h bf = ld16h(kp, kp + 8);
    v8f sc = {};
    sc = __builtin_amdgcn_wmma_f32_16x16x32_f16(false, qf, false, bf, (short)0, sc, false, false);
#pragma unroll
    for (int r = 0; r < 8; r++) {
      float s = sc[r] * scale;
      float tm = s;
      tm = fmaxf(tm, __shfl_xor(tm, 1));
      tm = fmaxf(tm, __shfl_xor(tm, 2));
      tm = fmaxf(tm, __shfl_xor(tm, 4));
      tm = fmaxf(tm, __shfl_xor(tm, 8));
      float mn = fmaxf(mrow[r], tm);
      float p = __expf(s - mn);
      float ts = p;
      ts += __shfl_xor(ts, 1);
      ts += __shfl_xor(ts, 2);
      ts += __shfl_xor(ts, 4);
      ts += __shfl_xor(ts, 8);
      srow[r] = srow[r] * __expf(mrow[r] - mn) + ts;
      mrow[r] = mn;
    }
  }
  float inv[8];
#pragma unroll
  for (int r = 0; r < 8; r++) inv[r] = 1.0f / srow[r];

  // ---- pass 2: write attn + accumulate ctx
  v8f cacc[2] = {};
  for (int k0 = 0; k0 < S_LEN; k0 += 32) {
#pragma unroll
    for (int t = 0; t < 2; t++) {
      const int kk = k0 + t * 16;
      const _Float16* kp = Kb + (size_t)(kk + l15) * D_LAT + kb;
      v16h bf = ld16h(kp, kp + 8);
      v8f sc = {};
      sc = __builtin_amdgcn_wmma_f32_16x16x32_f16(false, qf, false, bf, (short)0, sc, false, false);
      const int col = kk + l15;
#pragma unroll
      for (int r = 0; r < 8; r++) {
        float p = __expf(sc[r] * scale - mrow[r]) * inv[r];
        const int rowl = r + rloc;
        __builtin_nontemporal_store(p, attn_b + (size_t)(q0 + rowl) * S_LEN + col);
        pl[w][rowl][t * 16 + l15] = (_Float16)p;
      }
    }
    // relayout P (C-layout) -> A-fragment via per-wave LDS
    const _Float16* pp = &pl[w][l15][0];
    v16h pa = ld16h(pp + ks, pp + ks + 8);
#pragma unroll
    for (int ni = 0; ni < 2; ni++) {
      const _Float16* vp = Vb + (size_t)(ni * 16 + l15) * S_LEN + k0 + kb;
      v16h vf = ld16h(vp, vp + 8);
      cacc[ni] = __builtin_amdgcn_wmma_f32_16x16x32_f16(false, pa, false, vf, (short)0, cacc[ni], false, false);
    }
  }

#pragma unroll
  for (int ni = 0; ni < 2; ni++)
#pragma unroll
    for (int r = 0; r < 8; r++) {
      const int s = q0 + r + rloc;
      const int col = h * D_LAT + ni * 16 + l15;
      ctx[(size_t)(b * S_LEN + s) * (H_NUM * D_LAT) + col] = (_Float16)cacc[ni][r];
    }
}

// ---------------------------------------------------------------- launcher
extern "C" void kernel_launch(void* const* d_in, const int* in_sizes, int n_in,
                              void* d_out, int out_size, void* d_ws, size_t ws_size,
                              hipStream_t stream) {
  (void)in_sizes; (void)n_in; (void)out_size; (void)ws_size;

  const float* query = (const float*)d_in[0];
  const float* key_  = (const float*)d_in[1];
  const float* value = (const float*)d_in[2];
  const float* W_q  = (const float*)d_in[3];
  const float* b_q  = (const float*)d_in[4];
  const float* W_k  = (const float*)d_in[5];
  const float* b_k  = (const float*)d_in[6];
  const float* W_v  = (const float*)d_in[7];
  const float* b_v  = (const float*)d_in[8];
  const float* W_vo = (const float*)d_in[9];
  const float* b_vo = (const float*)d_in[10];
  const float* W_o  = (const float*)d_in[11];
  const float* b_o  = (const float*)d_in[12];

  char* ws = (char*)d_ws;
  size_t off = 0;
  auto alloc = [&](size_t bytes) -> void* {
    void* p = ws + off;
    off += (bytes + 255) & ~(size_t)255;
    return p;
  };

  const size_t nElem = (size_t)ROWS * 1024;               // 4,194,304
  _Float16* qh    = (_Float16*)alloc(nElem * 2);
  _Float16* kh    = (_Float16*)alloc(nElem * 2);
  _Float16* vh    = (_Float16*)alloc(nElem * 2);
  _Float16* WqT   = (_Float16*)alloc((size_t)512 * 1024 * 2);
  _Float16* WkT   = (_Float16*)alloc((size_t)512 * 1024 * 2);
  _Float16* WvT   = (_Float16*)alloc((size_t)512 * 1024 * 2);
  _Float16* WvoT  = (_Float16*)alloc((size_t)1024 * 512 * 2);
  _Float16* WoT   = (_Float16*)alloc((size_t)1024 * 1024 * 2);
  _Float16* Qlat  = (_Float16*)alloc((size_t)ROWS * 512 * 2);   // (B,H,S,32)
  _Float16* Klat  = (_Float16*)alloc((size_t)ROWS * 512 * 2);   // (B,H,S,32)
  _Float16* VlatT = (_Float16*)alloc((size_t)ROWS * 512 * 2);   // (B,H,32,S)
  _Float16* ctx   = (_Float16*)alloc((size_t)ROWS * 512 * 2);   // (B*S,512)
  _Float16* hidden= (_Float16*)alloc((size_t)ROWS * 1024 * 2);  // (B*S,1024)

  // f32 -> f16 inputs
  cvt_f32_f16<<<(int)(nElem / 1024), 256, 0, stream>>>(query, qh, (int)nElem);
  cvt_f32_f16<<<(int)(nElem / 1024), 256, 0, stream>>>(key_,  kh, (int)nElem);
  cvt_f32_f16<<<(int)(nElem / 1024), 256, 0, stream>>>(value, vh, (int)nElem);

  // weights -> (N,K) f16
  transpose_w<<<(512 * 1024) / 256, 256, 0, stream>>>(W_q,  WqT,  1024, 1024, 512, 1);
  transpose_w<<<(512 * 1024) / 256, 256, 0, stream>>>(W_k,  WkT,  1024, 512,  512, 0);
  transpose_w<<<(512 * 1024) / 256, 256, 0, stream>>>(W_v,  WvT,  1024, 512,  512, 0);
  transpose_w<<<(1024 * 512) / 256, 256, 0, stream>>>(W_vo, WvoT, 512,  1024, 1024, 0);
  transpose_w<<<(1024 * 1024) / 256, 256, 0, stream>>>(W_o, WoT,  1024, 1024, 1024, 0);

  // projections
  gemm_f16<<<dim3(ROWS / 64, 512 / 128), 256, 0, stream>>>(qh, WqT, b_q, Qlat,  ROWS, 512, 1024, 2, 1);
  gemm_f16<<<dim3(ROWS / 64, 512 / 128), 256, 0, stream>>>(kh, WkT, b_k, Klat,  ROWS, 512, 1024, 2, 0);
  gemm_f16<<<dim3(ROWS / 64, 512 / 128), 256, 0, stream>>>(vh, WvT, b_v, VlatT, ROWS, 512, 1024, 3, 0);

  // attention (writes attn f32 + ctx f16)
  float* out_p  = (float*)d_out;
  float* attn_p = out_p + (size_t)ROWS * 1024;
  mla_attention<<<dim3(S_LEN / 64, H_NUM, BATCH), 128, 0, stream>>>(Qlat, Klat, VlatT, ctx, attn_p);

  // output projections
  gemm_f16<<<dim3(ROWS / 64, 1024 / 128), 256, 0, stream>>>(ctx,    WvoT, b_vo, hidden, ROWS, 1024, 512,  1, 0);
  gemm_f16<<<dim3(ROWS / 64, 1024 / 128), 256, 0, stream>>>(hidden, WoT,  b_o,  out_p,  ROWS, 1024, 1024, 0, 0);
}
